// MultiHeadAttention_7352984010874
// MI455X (gfx1250) — matrix operand order repaired and checked by reading, not compiled, not run on hardware
//
#include <hip/hip_runtime.h>
#include <hip/hip_bf16.h>

// ---------------------------------------------------------------------------
// Multi-head attention forward for MI455X (gfx1250).
// bf16 WMMA pipeline + async global->LDS double-buffered staging (CDNA5).
// ---------------------------------------------------------------------------

typedef __attribute__((ext_vector_type(16))) __bf16 v16bf;
typedef __attribute__((ext_vector_type(8)))  __bf16 v8bf;
typedef __attribute__((ext_vector_type(8)))  float  v8f;

#define B_   2
#define S_   2048
#define DIM_ 1024
#define H_   16
#define D_   64

// --- WMMA helpers ----------------------------------------------------------

static __device__ inline v8f wmma_bf16(v16bf a, v16bf b, v8f c) {
  // D(16x16,f32) = A(16x32,bf16) * B(32x16,bf16) + C
  return __builtin_amdgcn_wmma_f32_16x16x32_bf16(
      /*neg_a=*/false, a, /*neg_b=*/false, b,
      /*c_mod=*/(short)0, c, /*reuse_a=*/false, /*reuse_b=*/false);
}

// A fragment (16x32 bf16): lane holds row m = lane&15; K chunk base
// koff = (lane>>4)*8 : elems 0..7 = K(koff..+7), elems 8..15 = K(koff+16..+23).
// `p` must already point at rowBase + koff.
static __device__ inline v16bf load_a16(const __bf16* p) {
  v16bf a;
#pragma unroll
  for (int i = 0; i < 8; ++i) { a[i] = p[i]; a[i + 8] = p[16 + i]; }
  return a;
}

// B fragment (32x16 bf16) from 16B-aligned memory: lane holds col n = lane&15,
// koff = (lane>>4)*8: elems 0..7 = K(koff..+7), elems 8..15 = K(koff+16..+23).
static __device__ inline v16bf load_b16(const __bf16* p) {
  v8bf x = *(const v8bf*)p;
  v8bf y = *(const v8bf*)(p + 16);
  v16bf r;
#pragma unroll
  for (int i = 0; i < 8; ++i) { r[i] = x[i]; r[i + 8] = y[i]; }
  return r;
}

// --- CDNA5 async global->LDS copy (ASYNCcnt-tracked) -----------------------
// Generic pointers to LDS carry the LDS byte offset in bits [31:0] (aperture
// scheme, ISA 10.2), so the low dword of a __shared__ pointer is a valid VDST.

static __device__ inline void async_b128(unsigned lds_off, const void* gptr) {
  asm volatile("global_load_async_to_lds_b128 %0, %1, off"
               :: "v"(lds_off), "v"(gptr) : "memory");
}
static __device__ inline void wait_async_le(int n) {  // n must be literal-like
  if (n == 0) asm volatile("s_wait_asynccnt 0x0" ::: "memory");
  else if (n == 2) asm volatile("s_wait_asynccnt 0x2" ::: "memory");
  else asm volatile("s_wait_asynccnt 0x4" ::: "memory");
}

// ---------------------------------------------------------------------------
// Elementwise prep kernels
// ---------------------------------------------------------------------------

__global__ void f32_to_bf16_kernel(const float* __restrict__ src,
                                   __bf16* __restrict__ dst, int n) {
  int i = blockIdx.x * blockDim.x + threadIdx.x;
  if (i < n) dst[i] = (__bf16)src[i];
}

// W[K][N] f32 (row-major, used as X@W) -> Wt[N][K] bf16 (K contiguous per row)
__global__ void wtrans_kernel(const float* __restrict__ W,
                              __bf16* __restrict__ Wt) {
  int i = blockIdx.x * blockDim.x + threadIdx.x;  // 1024*1024
  if (i >= DIM_ * DIM_) return;
  int n = i & (DIM_ - 1);
  int k = i >> 10;
  Wt[(size_t)n * DIM_ + k] = (__bf16)W[(size_t)k * DIM_ + n];
}

// RoPE: P[B,S,DIM] bf16 -> O[B,H,S,D] bf16; `scale` folds 1/sqrt(D) into Q.
__global__ void rope_kernel(const __bf16* __restrict__ P,
                            __bf16* __restrict__ O, float scale) {
  int i = blockIdx.x * blockDim.x + threadIdx.x;  // B*S*H*32 = 2^21
  if (i >= B_ * S_ * H_ * (D_ / 2)) return;
  int d2 = i & 31;
  int h  = (i >> 5) & (H_ - 1);
  int s  = (i >> 9) & (S_ - 1);
  int b  = i >> 20;
  float inv_freq = __powf(10000.0f, -((float)(2 * d2)) / (float)D_);
  float ang = (float)s * inv_freq;
  float sn, cs;
  __sincosf(ang, &sn, &cs);
  const __bf16* src = P + ((size_t)(b * S_ + s)) * DIM_ + h * D_ + 2 * d2;
  float x1 = (float)src[0], x2 = (float)src[1];
  __bf16* dst = O + (((size_t)(b * H_ + h) * S_) + s) * D_ + 2 * d2;
  dst[0] = (__bf16)((x1 * cs - x2 * sn) * scale);
  dst[1] = (__bf16)((x1 * sn + x2 * cs) * scale);
}

// ---------------------------------------------------------------------------
// bf16 WMMA GEMM: one 16(M) x 64(N) macro-tile per wave: the A fragment is
// reused by 4 WMMAs per K-step (4x lower A traffic than 16x16 tiles).
// MODE 0: store bf16 row-major [M,N]
// MODE 1: store bf16 head-transposed Vt[B,H,D,S]
// MODE 2: store f32 [M,N] with bias[n] (final output)
// ---------------------------------------------------------------------------

template <int MODE>
__global__ __launch_bounds__(256) void gemm_bf16_kernel(
    const __bf16* __restrict__ X, const __bf16* __restrict__ Wt,
    void* __restrict__ outp, const float* __restrict__ bias,
    int M, int N, int K) {
  const int lane = threadIdx.x & 31;
  const int l = lane & 15, hi = lane >> 4;
  const int wid = blockIdx.x * 8 + (threadIdx.x >> 5);
  const int ncols = N >> 6;                       // 64-wide N groups
  if (wid >= (M >> 4) * ncols) return;            // wave-uniform exit
  const int m0 = (wid / ncols) << 4;
  const int n0 = (wid % ncols) << 6;

  const __bf16* arow = X + (size_t)(m0 + l) * K + hi * 8;
  const __bf16* brow0 = Wt + (size_t)(n0 + l) * K + hi * 8;

  v8f acc[4] = {};
#pragma unroll 2
  for (int kb = 0; kb < K; kb += 32) {
    v16bf a = load_a16(arow + kb);
#pragma unroll
    for (int c = 0; c < 4; ++c) {
      v16bf b = load_b16(brow0 + (size_t)(c * 16) * K + kb);
      acc[c] = wmma_bf16(a, b, acc[c]);
    }
  }

#pragma unroll
  for (int c = 0; c < 4; ++c) {
    const int n = n0 + c * 16 + l;
    if (MODE == 0) {
      __bf16* out = (__bf16*)outp;
#pragma unroll
      for (int r = 0; r < 8; ++r)
        out[(size_t)(m0 + hi * 8 + r) * N + n] = (__bf16)acc[c][r];
    } else if (MODE == 1) {
      __bf16* out = (__bf16*)outp;
      const int h = n >> 6, d = n & 63;
      const int b = m0 / S_;
      const int s0 = (m0 - b * S_) + hi * 8;
      v8bf v;
#pragma unroll
      for (int r = 0; r < 8; ++r) v[r] = (__bf16)acc[c][r];
      *(v8bf*)(out + (((size_t)(b * H_ + h) * D_) + d) * S_ + s0) = v;
    } else {
      float* out = (float*)outp;
      const float bn = bias[n];
#pragma unroll
      for (int r = 0; r < 8; ++r)
        out[(size_t)(m0 + hi * 8 + r) * N + n] = acc[c][r] + bn;
    }
  }
}

// ---------------------------------------------------------------------------
// Attention core: one block = one (b, h, 16-query tile). 128 threads = 4 waves.
// LDS: 16x2048 f32 scores (131 KB) + per-wave double-buffered async stage
// buffers for K (2x 2.25 KB) and V (2x 1.25 KB). Total ~156 KB -> 2 blocks/WGP.
// ---------------------------------------------------------------------------

#define SPAD   (S_ + 4)  // break 64-bank alignment of 2048-f32 rows
#define KROW   72        // 64 elems + 8 pad (144 B rows, conflict-free reads)
#define VROW   40        // 32 elems + 8 pad (80 B rows)

__global__ __launch_bounds__(128) void attn_kernel(
    const __bf16* __restrict__ Q,   // [B,H,S,D] (roped, pre-scaled by 1/8)
    const __bf16* __restrict__ Kt,  // [B,H,S,D] (roped)
    const __bf16* __restrict__ Vt,  // [B,H,D,S]
    const int*    __restrict__ mask,  // [B,1,S,S]
    __bf16* __restrict__ AO)        // [B,S,DIM]
{
  __shared__ float  sS[16][SPAD];
  __shared__ float  sSum[16];
  __shared__ __bf16 kst[4][2][16 * KROW];
  __shared__ __bf16 vst[4][2][16 * VROW];

  const int qt = blockIdx.x & 127;
  const int h  = (blockIdx.x >> 7) & (H_ - 1);
  const int b  = blockIdx.x >> 11;
  const int q0 = qt << 4;
  const int tid = threadIdx.x;
  const int w = tid >> 5, lane = tid & 31, l = lane & 15, hi = lane >> 4;

  const __bf16* Qbase = Q  + ((size_t)(b * H_ + h) * S_ + q0) * D_;
  const __bf16* Kbase = Kt + (size_t)(b * H_ + h) * S_ * D_;
  const __bf16* Vbase = Vt + (size_t)(b * H_ + h) * D_ * S_;

  // ---- Phase 1: scores = Q Kt^T (1/8 pre-folded into Q). ----
  // Q A-fragments loaded once and reused over all 32 key tiles of this wave.
  const __bf16* qrow = Qbase + (size_t)l * D_ + hi * 8;
  const v16bf qa0 = load_a16(qrow);       // head dims 0..31
  const v16bf qa1 = load_a16(qrow + 32);  // head dims 32..63

  // async stage of one 16x64 K tile (2 KB contiguous): 4 b128 issues/lane.
  auto stage_k = [&](int jj, int buf) {
    unsigned lb = (unsigned)(uintptr_t)&kst[w][buf][0];
    const char* g = (const char*)(Kbase + (size_t)jj * 16 * D_);
#pragma unroll
    for (int q = 0; q < 4; ++q) {
      int c = q * 32 + lane;        // 128 x 16B chunks
      int r = c >> 3, o = (c & 7) * 16;
      async_b128(lb + (unsigned)(r * KROW * 2 + o), g + r * 128 + o);
    }
  };

  stage_k(w * 32, 0);
  for (int t = 0; t < 32; ++t) {                  // uniform trip count
    stage_k(w * 32 + ((t + 1) & 31), (t + 1) & 1);
    wait_async_le(4);                             // buffer t landed
    const __bf16* kb = &kst[w][t & 1][0] + l * KROW + hi * 8;
    v16bf b0 = load_b16(kb);
    v16bf b1 = load_b16(kb + 32);
    v8f acc = {};
    acc = wmma_bf16(qa0, b0, acc);
    acc = wmma_bf16(qa1, b1, acc);
    const int k0 = (w * 32 + t) << 4;
#pragma unroll
    for (int r = 0; r < 8; ++r)
      sS[hi * 8 + r][k0 + l] = acc[r];
  }
  __syncthreads();

  // ---- Phase 2: masked softmax (unnormalized exp left in LDS) ----
  {
    const int r  = tid >> 3;          // 8 lanes per row, same wave
    const int c0 = (tid & 7) << 8;    // 256 keys per lane
    const int* mrow = mask + ((size_t)b * S_ + (q0 + r)) * S_;
    float mx = -3.0e38f;
    for (int c = c0; c < c0 + 256; c += 4) {
      int4 mm = *(const int4*)(mrow + c);
      float4 sv = *(const float4*)&sS[r][c];
      sv.x = mm.x ? sv.x : -10000.0f;
      sv.y = mm.y ? sv.y : -10000.0f;
      sv.z = mm.z ? sv.z : -10000.0f;
      sv.w = mm.w ? sv.w : -10000.0f;
      *(float4*)&sS[r][c] = sv;
      mx = fmaxf(mx, fmaxf(fmaxf(sv.x, sv.y), fmaxf(sv.z, sv.w)));
    }
    mx = fmaxf(mx, __shfl_xor(mx, 1));
    mx = fmaxf(mx, __shfl_xor(mx, 2));
    mx = fmaxf(mx, __shfl_xor(mx, 4));
    float sum = 0.0f;
    for (int c = c0; c < c0 + 256; c += 4) {
      float4 sv = *(const float4*)&sS[r][c];
      sv.x = __expf(sv.x - mx);
      sv.y = __expf(sv.y - mx);
      sv.z = __expf(sv.z - mx);
      sv.w = __expf(sv.w - mx);
      *(float4*)&sS[r][c] = sv;
      sum += (sv.x + sv.y) + (sv.z + sv.w);
    }
    sum += __shfl_xor(sum, 1);
    sum += __shfl_xor(sum, 2);
    sum += __shfl_xor(sum, 4);
    if ((tid & 7) == 0) sSum[r] = sum;
  }
  __syncthreads();

  // ---- Phase 3: O = (P V) / rowsum. Wave w owns d-chunk n0 = 16*w. ----
  {
    const int n0 = w << 4;
    // async stage of one 16(d) x 32(key) V window (1 KB): 2 b128 issues/lane.
    auto stage_v = [&](int kb, int buf) {
      unsigned lb = (unsigned)(uintptr_t)&vst[w][buf][0];
      const __bf16* g = Vbase + (size_t)n0 * S_ + kb;
#pragma unroll
      for (int q = 0; q < 2; ++q) {
        int c = q * 32 + lane;        // 64 x 16B chunks
        int r = c & 15, half = c >> 4;
        async_b128(lb + (unsigned)(r * VROW * 2 + half * 16),
                   (const char*)g + ((size_t)r * S_ + half * 8) * 2);
      }
    };

    stage_v(0, 0);
    v8f acc = {};
    for (int t = 0; t < 64; ++t) {                // uniform trip count
      stage_v(((t + 1) & 63) * 32, (t + 1) & 1);
      wait_async_le(2);
      const int kb = t * 32;
      const float* prow = &sS[l][kb + hi * 8];
      v16bf a;
#pragma unroll
      for (int i = 0; i < 8; ++i) {
        a[i]     = (__bf16)prow[i];
        a[i + 8] = (__bf16)prow[16 + i];
      }
      v16bf bf = load_b16(&vst[w][t & 1][0] + l * VROW + hi * 8);
      acc = wmma_bf16(a, bf, acc);
    }
#pragma unroll
    for (int r = 0; r < 8; ++r) {
      const int m = hi * 8 + r;
      float val = acc[r] / sSum[m];
      AO[(size_t)(b * S_ + q0 + m) * DIM_ + h * D_ + n0 + l] = (__bf16)val;
    }
  }
}

// ---------------------------------------------------------------------------
// Host-side orchestration
// ---------------------------------------------------------------------------

extern "C" void kernel_launch(void* const* d_in, const int* in_sizes, int n_in,
                              void* d_out, int out_size, void* d_ws, size_t ws_size,
                              hipStream_t stream) {
  (void)in_sizes; (void)n_in; (void)out_size; (void)ws_size;

  const float* q   = (const float*)d_in[0];
  const float* k   = (const float*)d_in[1];
  const float* v   = (const float*)d_in[2];
  const int*   msk = (const int*)  d_in[3];
  const float* Wq  = (const float*)d_in[4];
  const float* Wk  = (const float*)d_in[5];
  const float* Wv  = (const float*)d_in[6];
  const float* Wo  = (const float*)d_in[7];
  const float* bo  = (const float*)d_in[8];
  float* out = (float*)d_out;

  const size_t NELEM = (size_t)B_ * S_ * DIM_;  // 4 Mi
  char* w = (char*)d_ws;
  __bf16* Xq  = (__bf16*)w; w += NELEM * 2;
  __bf16* Xk  = (__bf16*)w; w += NELEM * 2;
  __bf16* Xv  = (__bf16*)w; w += NELEM * 2;
  __bf16* WtQ = (__bf16*)w; w += (size_t)DIM_ * DIM_ * 2;
  __bf16* WtK = (__bf16*)w; w += (size_t)DIM_ * DIM_ * 2;
  __bf16* WtV = (__bf16*)w; w += (size_t)DIM_ * DIM_ * 2;
  __bf16* WtO = (__bf16*)w; w += (size_t)DIM_ * DIM_ * 2;
  __bf16* Qp  = (__bf16*)w; w += NELEM * 2;   // pre-rope proj [B,S,DIM]
  __bf16* Kp  = (__bf16*)w; w += NELEM * 2;
  __bf16* Qr  = (__bf16*)w; w += NELEM * 2;   // roped [B,H,S,D]
  __bf16* Kr  = (__bf16*)w; w += NELEM * 2;
  __bf16* Vt  = (__bf16*)w; w += NELEM * 2;   // [B,H,D,S]
  __bf16* AO  = (__bf16*)w; w += NELEM * 2;   // attention out [B,S,DIM]

  const int cvt_grid = (int)(NELEM / 256);
  f32_to_bf16_kernel<<<cvt_grid, 256, 0, stream>>>(q, Xq, (int)NELEM);
  f32_to_bf16_kernel<<<cvt_grid, 256, 0, stream>>>(k, Xk, (int)NELEM);
  f32_to_bf16_kernel<<<cvt_grid, 256, 0, stream>>>(v, Xv, (int)NELEM);

  const int wt_grid = (DIM_ * DIM_) / 256;
  wtrans_kernel<<<wt_grid, 256, 0, stream>>>(Wq, WtQ);
  wtrans_kernel<<<wt_grid, 256, 0, stream>>>(Wk, WtK);
  wtrans_kernel<<<wt_grid, 256, 0, stream>>>(Wv, WtV);
  wtrans_kernel<<<wt_grid, 256, 0, stream>>>(Wo, WtO);

  const int M = B_ * S_;                         // 4096
  const int tiles = (M / 16) * (DIM_ / 64);      // 4096 wave-tiles (16x64)
  const int ggrid = (tiles + 7) / 8;             // 8 waves / 256-thread block
  gemm_bf16_kernel<0><<<ggrid, 256, 0, stream>>>(Xq, WtQ, Qp, nullptr, M, DIM_, DIM_);
  gemm_bf16_kernel<0><<<ggrid, 256, 0, stream>>>(Xk, WtK, Kp, nullptr, M, DIM_, DIM_);
  gemm_bf16_kernel<1><<<ggrid, 256, 0, stream>>>(Xv, WtV, Vt, nullptr, M, DIM_, DIM_);

  const int rope_grid = (B_ * S_ * H_ * (D_ / 2)) / 256;  // 8192
  rope_kernel<<<rope_grid, 256, 0, stream>>>(Qp, Qr, 0.125f);  // 1/sqrt(64)
  rope_kernel<<<rope_grid, 256, 0, stream>>>(Kp, Kr, 1.0f);

  attn_kernel<<<B_ * H_ * (S_ / 16), 128, 0, stream>>>(Qr, Kr, Vt, msk, AO);

  gemm_bf16_kernel<2><<<ggrid, 256, 0, stream>>>(AO, WtO, out, bo, M, DIM_, DIM_);
}
